// RelativeTransformerEncoderLayer_29231547417060
// MI455X (gfx1250) — matrix operand order repaired and checked by reading, not compiled, not run on hardware
//
#include <hip/hip_runtime.h>
#include <hip/hip_bf16.h>
#include <stdint.h>

#define DEV __device__ __forceinline__

typedef __attribute__((ext_vector_type(16))) __bf16 v16bf;
typedef __attribute__((ext_vector_type(8)))  float  v8f;
typedef __attribute__((ext_vector_type(4)))  unsigned v4u;
typedef __attribute__((ext_vector_type(8)))  int      v8i;
typedef __attribute__((ext_vector_type(4)))  int      v4i;

#if defined(__HIP_DEVICE_COMPILE__) && __has_builtin(__builtin_amdgcn_tensor_load_to_lds)
#define ATHENA_TDM 1
#endif

union FragB { v16bf v; unsigned short u[16]; };

DEV unsigned short f2bf(float f) {
  unsigned int u = __builtin_bit_cast(unsigned int, f);
  u += 0x7FFFu + ((u >> 16) & 1u);           // round-to-nearest-even
  return (unsigned short)(u >> 16);
}
DEV unsigned int pack2bf(float a, float b) {
  return (unsigned int)f2bf(a) | ((unsigned int)f2bf(b) << 16);
}
DEV float gelu_tanh(float x) {
  const float kC = 0.7978845608028654f;      // sqrt(2/pi)
  return 0.5f * x * (1.0f + tanhf(kC * (x + 0.044715f * x * x * x)));
}

#ifdef ATHENA_TDM
// 2D TDM descriptor load: tile (tile_k x tile_rows) of bf16 from a row-major
// [rows x dim_k] tensor (row stride = stride_k elements) into LDS at lds_off,
// padded +8 DWORDs every 16 DWORDs => LDS row stride 48 ushorts.
DEV void tdm_load_tile_2d(unsigned lds_off, const unsigned short* gptr,
                          unsigned tile_k, unsigned tile_rows,
                          unsigned dim_k, unsigned dim_rows,
                          unsigned long long stride_k) {
  unsigned long long ga = (unsigned long long)(uintptr_t)gptr;
  v4u g0;
  g0[0] = 1u;                                            // count=1 (valid user D#)
  g0[1] = lds_off;                                       // lds_addr
  g0[2] = (unsigned)(ga & 0xFFFFFFFFu);                  // global_addr[31:0]
  g0[3] = (unsigned)((ga >> 32) & 0x01FFFFFFu) | 0x80000000u; // addr[56:32] | type=2
  v8i g1;
  // data_size=1(2B)@16, pad_enable@20, pad_interval=3(16 dw)@22, pad_amount=7(8 dw)@25
  g1[0] = (int)((1u << 16) | (1u << 20) | (3u << 22) | (7u << 25));
  g1[1] = (int)((dim_k & 0xFFFFu) << 16);                              // dim0[15:0]
  g1[2] = (int)(((dim_k >> 16) & 0xFFFFu) | ((dim_rows & 0xFFFFu) << 16)); // dim0 hi | dim1 lo
  g1[3] = (int)(((dim_rows >> 16) & 0xFFFFu) | ((tile_k & 0xFFFFu) << 16)); // dim1 hi | tile_dim0
  g1[4] = (int)(tile_rows & 0xFFFFu);                                  // tile_dim1 (tile_dim2=0)
  g1[5] = (int)(stride_k & 0xFFFFFFFFull);                             // stride0 lo
  g1[6] = (int)((stride_k >> 32) & 0xFFFFull);                         // stride0 hi
  g1[7] = 0;
  v4i g2 = {0, 0, 0, 0}, g3 = {0, 0, 0, 0};
  v8i g4 = {0, 0, 0, 0, 0, 0, 0, 0};
  __builtin_amdgcn_tensor_load_to_lds(g0, g1, g2, g3, g4, 0);
}
#endif

// ---------------- fp32 -> bf16 bulk conversion ----------------
__global__ __launch_bounds__(256) void cvt_bf16_kernel(const float* __restrict__ in,
                                                       unsigned short* __restrict__ out,
                                                       int n4) {
  int i = blockIdx.x * 256 + threadIdx.x;
  if (i >= n4) return;
  float4 f = reinterpret_cast<const float4*>(in)[i];
  uint2 p;
  p.x = pack2bf(f.x, f.y);
  p.y = pack2bf(f.z, f.w);
  reinterpret_cast<uint2*>(out)[i] = p;
}

// ---------------- fp32 [R,C] -> transposed bf16 [C,R] (tiled) ----------------
__global__ __launch_bounds__(256) void cvt_t_bf16_kernel(const float* __restrict__ in,
                                                         unsigned short* __restrict__ out,
                                                         int R, int C) {
  __shared__ unsigned short tile[32][33];
  const int bx = blockIdx.x * 32;  // col tile
  const int by = blockIdx.y * 32;  // row tile
  const int tx = threadIdx.x & 31, ty = threadIdx.x >> 5;  // 32 x 8
#pragma unroll
  for (int i = 0; i < 32; i += 8)
    tile[ty + i][tx] = f2bf(in[(size_t)(by + ty + i) * C + bx + tx]);
  __syncthreads();
#pragma unroll
  for (int i = 0; i < 32; i += 8)
    out[(size_t)(bx + ty + i) * R + by + tx] = tile[tx][ty + i];
}

// ---------------- bf16 WMMA GEMM: C = A[M,K] * Bt[N,K]^T + bias ----------------
// block = 256 threads (8 waves). Block tile 128x64, wave tile 16x64, K chunk 32.
// Both A and B tiles K-contiguous in LDS (48-ushort padded rows).
__global__ __launch_bounds__(256) void gemm_bf16_wmma(
    const unsigned short* __restrict__ A,   // [M,K] bf16 row-major
    const unsigned short* __restrict__ Bt,  // [N,K] bf16 row-major (W transposed)
    const float* __restrict__ bias, float* __restrict__ Cf,
    unsigned short* __restrict__ Cb, int M, int N, int K, int do_gelu) {
  __shared__ __align__(16) unsigned short As[128 * 48];
  __shared__ __align__(16) unsigned short Bs[64 * 48];

  const int t = threadIdx.x;
  const int wave = t >> 5, lane = t & 31;
  const int half = lane >> 4, l15 = lane & 15;
  const int mBlock = blockIdx.y * 128;
  const int nBlock = blockIdx.x * 64;

  v8f acc[4];
#pragma unroll
  for (int nt = 0; nt < 4; ++nt)
#pragma unroll
    for (int j = 0; j < 8; ++j) acc[nt][j] = 0.0f;

  for (int k0 = 0; k0 < K; k0 += 32) {
#ifdef ATHENA_TDM
    if (wave == 0) {
      tdm_load_tile_2d((unsigned)(uintptr_t)&As[0],
                       A + (size_t)mBlock * K + k0, 32u, 128u,
                       (unsigned)K, (unsigned)M, (unsigned long long)K);
      tdm_load_tile_2d((unsigned)(uintptr_t)&Bs[0],
                       Bt + (size_t)nBlock * K + k0, 32u, 64u,
                       (unsigned)K, (unsigned)N, (unsigned long long)K);
      __builtin_amdgcn_s_wait_tensorcnt((short)0);
    }
#else
    for (int c = t; c < 512; c += 256) {         // A tile 128x32
      int row = c >> 2, col8 = (c & 3) << 3;
      *reinterpret_cast<uint4*>(&As[row * 48 + col8]) =
          *reinterpret_cast<const uint4*>(A + (size_t)(mBlock + row) * K + k0 + col8);
    }
    {                                            // B tile 64x32 (K-contiguous rows)
      int row = t >> 2, col8 = (t & 3) << 3;
      *reinterpret_cast<uint4*>(&Bs[row * 48 + col8]) =
          *reinterpret_cast<const uint4*>(Bt + (size_t)(nBlock + row) * K + k0 + col8);
    }
    if (k0 + 32 < K) {
      __builtin_prefetch(A + (size_t)(mBlock + (t >> 1)) * K + k0 + 32, 0, 0);
      __builtin_prefetch(Bt + (size_t)(nBlock + (t & 63)) * K + k0 + 32, 0, 0);
    }
#endif
    __syncthreads();

    // A fragment: lanes<16 K=0..15, lanes>=16 K=16..31
    FragB fa;
    const int arow = wave * 16 + l15;
#pragma unroll
    for (int j = 0; j < 16; ++j) {
      int kk = j + half * 16;
      fa.u[j] = As[arow * 48 + kk];
    }
#pragma unroll
    for (int nt = 0; nt < 4; ++nt) {
      // B fragment: B(kk,n) = Bt[n][kk]; lanes<16 K=0..15, lanes>=16 K=16..31
      FragB fb;
#pragma unroll
      for (int j = 0; j < 16; ++j)
        fb.u[j] = Bs[(nt * 16 + l15) * 48 + j + half * 16];
      acc[nt] = __builtin_amdgcn_wmma_f32_16x16x32_bf16(
          false, fa.v, false, fb.v, (short)0, acc[nt], false, false);
    }
    __syncthreads();
  }

#pragma unroll
  for (int nt = 0; nt < 4; ++nt) {
    int col = nBlock + nt * 16 + l15;
    float bb = bias[col];
#pragma unroll
    for (int j = 0; j < 8; ++j) {
      int row = mBlock + wave * 16 + half * 8 + j;
      float vv = acc[nt][j] + bb;
      if (do_gelu) vv = gelu_tanh(vv);
      if (Cf) Cf[(size_t)row * N + col] = vv;
      if (Cb) Cb[(size_t)row * N + col] = f2bf(vv);
    }
  }
}

// ---------------- attention: one block per (b,h); L=64, dh=64 ----------------
__global__ __launch_bounds__(256) void attn_wmma_kernel(
    const float* __restrict__ q, const float* __restrict__ k,
    const float* __restrict__ v, const float* __restrict__ relb,
    unsigned short* __restrict__ ctxb) {
  __shared__ __align__(16) unsigned short qs[64 * 72];  // q bf16, reused for probs
  __shared__ __align__(16) unsigned short ks[64 * 72];
  __shared__ __align__(16) unsigned short vT[64 * 72];  // v transposed: vT[d][l]
  __shared__ float sc[64 * 68];
  __shared__ float red[64][4];
  __shared__ float rowstat[64];

  const int t = threadIdx.x;
  const int wave = t >> 5, lane = t & 31;
  const int half = lane >> 4, l15 = lane & 15;
  const int b = blockIdx.x >> 4, h = blockIdx.x & 15;
  const size_t base = ((size_t)b * 64) * 1024 + (size_t)h * 64;

  for (int c = t; c < 1024; c += 256) {
    int l = c >> 4, d4 = (c & 15) << 2;
    size_t off = base + (size_t)l * 1024 + d4;
    float4 fq = *reinterpret_cast<const float4*>(q + off);
    float4 fk = *reinterpret_cast<const float4*>(k + off);
    float4 fv = *reinterpret_cast<const float4*>(v + off);
    uint2 pq = { pack2bf(fq.x, fq.y), pack2bf(fq.z, fq.w) };
    uint2 pk = { pack2bf(fk.x, fk.y), pack2bf(fk.z, fk.w) };
    *reinterpret_cast<uint2*>(&qs[l * 72 + d4]) = pq;
    *reinterpret_cast<uint2*>(&ks[l * 72 + d4]) = pk;
    vT[(d4 + 0) * 72 + l] = f2bf(fv.x);
    vT[(d4 + 1) * 72 + l] = f2bf(fv.y);
    vT[(d4 + 2) * 72 + l] = f2bf(fv.z);
    vT[(d4 + 3) * 72 + l] = f2bf(fv.w);
  }
  __syncthreads();

  // scores = q k^T / sqrt(dh) + rel_bias  (16 tiles of 16x16, 2 per wave)
  for (int tt = wave; tt < 16; tt += 8) {
    int mt = tt >> 2, nt = tt & 3;
    v8f acc;
#pragma unroll
    for (int j = 0; j < 8; ++j) acc[j] = 0.0f;
    for (int kc = 0; kc < 64; kc += 32) {
      FragB fa, fb;
#pragma unroll
      for (int j = 0; j < 16; ++j)
        fa.u[j] = qs[(mt * 16 + l15) * 72 + kc + j + half * 16];
#pragma unroll
      for (int j = 0; j < 16; ++j)   // B = k^T : element(kk,n) = k[n][kk]  (contiguous)
        fb.u[j] = ks[(nt * 16 + l15) * 72 + kc + j + half * 16];
      acc = __builtin_amdgcn_wmma_f32_16x16x32_bf16(
          false, fa.v, false, fb.v, (short)0, acc, false, false);
    }
#pragma unroll
    for (int j = 0; j < 8; ++j) {
      int m = mt * 16 + half * 8 + j, n = nt * 16 + l15;
      sc[m * 68 + n] = acc[j] * 0.125f + relb[h * 127 + m - n + 63];
    }
  }
  __syncthreads();

  // row softmax (4 lanes per row, 16 cols each)
  {
    int r = t >> 2, qq = t & 3, c0 = qq * 16;
    float mx = -3.4e38f;
    for (int c = 0; c < 16; ++c) mx = fmaxf(mx, sc[r * 68 + c0 + c]);
    red[r][qq] = mx;
    __syncthreads();
    if (t < 64)
      rowstat[t] = fmaxf(fmaxf(red[t][0], red[t][1]), fmaxf(red[t][2], red[t][3]));
    __syncthreads();
    float rm = rowstat[r], s = 0.f;
    for (int c = 0; c < 16; ++c) {
      float e = __expf(sc[r * 68 + c0 + c] - rm);
      sc[r * 68 + c0 + c] = e;
      s += e;
    }
    __syncthreads();
    red[r][qq] = s;
    __syncthreads();
    if (t < 64) rowstat[t] = 1.0f / (red[t][0] + red[t][1] + red[t][2] + red[t][3]);
    __syncthreads();
    float inv = rowstat[r];
    for (int c = 0; c < 16; ++c)
      qs[r * 72 + c0 + c] = f2bf(sc[r * 68 + c0 + c] * inv);  // probs -> qs (bf16)
  }
  __syncthreads();

  // ctx = probs @ v : B(kk,n) = v[kk][n] = vT[n][kk]  (contiguous reads)
  for (int tt = wave; tt < 16; tt += 8) {
    int mt = tt >> 2, nt = tt & 3;
    v8f acc;
#pragma unroll
    for (int j = 0; j < 8; ++j) acc[j] = 0.0f;
    for (int kc = 0; kc < 64; kc += 32) {
      FragB fa, fb;
#pragma unroll
      for (int j = 0; j < 16; ++j)
        fa.u[j] = qs[(mt * 16 + l15) * 72 + kc + j + half * 16];
#pragma unroll
      for (int j = 0; j < 16; ++j)
        fb.u[j] = vT[(nt * 16 + l15) * 72 + kc + j + half * 16];
      acc = __builtin_amdgcn_wmma_f32_16x16x32_bf16(
          false, fa.v, false, fb.v, (short)0, acc, false, false);
    }
#pragma unroll
    for (int j = 0; j < 8; ++j) {
      int l = mt * 16 + half * 8 + j, d = nt * 16 + l15;
      ctxb[((size_t)(b * 64 + l)) * 1024 + h * 64 + d] = f2bf(acc[j]);
    }
  }
}

// ---------------- residual + LayerNorm over D=1024, one block per row ----------------
__global__ __launch_bounds__(256) void ln_kernel(
    const float* __restrict__ a, const float* __restrict__ bres,
    const float* __restrict__ g, const float* __restrict__ be,
    float* __restrict__ outF, unsigned short* __restrict__ outB) {
  __shared__ float red[256];
  __shared__ float stat;
  const int t = threadIdx.x;
  const size_t row = (size_t)blockIdx.x * 1024;

  float4 va = *reinterpret_cast<const float4*>(a + row + t * 4);
  float4 vb = *reinterpret_cast<const float4*>(bres + row + t * 4);
  float x0 = va.x + vb.x, x1 = va.y + vb.y, x2 = va.z + vb.z, x3 = va.w + vb.w;

  red[t] = x0 + x1 + x2 + x3;
  __syncthreads();
  for (int o = 128; o > 0; o >>= 1) { if (t < o) red[t] += red[t + o]; __syncthreads(); }
  if (t == 0) stat = red[0] * (1.0f / 1024.0f);
  __syncthreads();
  float mu = stat;
  float d0 = x0 - mu, d1 = x1 - mu, d2 = x2 - mu, d3 = x3 - mu;
  __syncthreads();
  red[t] = d0 * d0 + d1 * d1 + d2 * d2 + d3 * d3;
  __syncthreads();
  for (int o = 128; o > 0; o >>= 1) { if (t < o) red[t] += red[t + o]; __syncthreads(); }
  if (t == 0) stat = rsqrtf(red[0] * (1.0f / 1024.0f) + 1e-5f);
  __syncthreads();
  float inv = stat;

  float4 gg = *reinterpret_cast<const float4*>(g + t * 4);
  float4 bb = *reinterpret_cast<const float4*>(be + t * 4);
  float4 y = { d0 * inv * gg.x + bb.x, d1 * inv * gg.y + bb.y,
               d2 * inv * gg.z + bb.z, d3 * inv * gg.w + bb.w };
  *reinterpret_cast<float4*>(outF + row + t * 4) = y;
  if (outB) {
    uint2 p = { pack2bf(y.x, y.y), pack2bf(y.z, y.w) };
    *reinterpret_cast<uint2*>(outB + row + t * 4) = p;
  }
}

// ---------------- host side ----------------
extern "C" void kernel_launch(void* const* d_in, const int* in_sizes, int n_in,
                              void* d_out, int out_size, void* d_ws, size_t ws_size,
                              hipStream_t stream) {
  const float* src  = (const float*)d_in[0];
  const float* Wq   = (const float*)d_in[1];
  const float* bq   = (const float*)d_in[2];
  const float* Wk   = (const float*)d_in[3];
  const float* bk   = (const float*)d_in[4];
  const float* Wv   = (const float*)d_in[5];
  const float* bv   = (const float*)d_in[6];
  const float* Wo   = (const float*)d_in[7];
  const float* bo   = (const float*)d_in[8];
  const float* relb = (const float*)d_in[9];
  const float* W1   = (const float*)d_in[10];
  const float* b1   = (const float*)d_in[11];
  const float* W2   = (const float*)d_in[12];
  const float* b2   = (const float*)d_in[13];
  const float* g1   = (const float*)d_in[14];
  const float* be1  = (const float*)d_in[15];
  const float* g2   = (const float*)d_in[16];
  const float* be2  = (const float*)d_in[17];
  float* out = (float*)d_out;

  const int BL = 256 * 64;  // 16384 rows
  const int D = 1024, F = 4096;

  char* w = (char*)d_ws;
  auto take = [&](size_t bytes) -> char* {
    char* p = w;
    w += (bytes + 255) & ~(size_t)255;
    return p;
  };

  unsigned short* srcB = (unsigned short*)take((size_t)BL * D * 2);
  unsigned short* WqT  = (unsigned short*)take((size_t)D * D * 2);   // [N,K]
  unsigned short* WkT  = (unsigned short*)take((size_t)D * D * 2);
  unsigned short* WvT  = (unsigned short*)take((size_t)D * D * 2);
  unsigned short* WoT  = (unsigned short*)take((size_t)D * D * 2);
  unsigned short* W1T  = (unsigned short*)take((size_t)D * F * 2);   // [F,D]
  unsigned short* W2T  = (unsigned short*)take((size_t)F * D * 2);   // [D,F]
  float* qf   = (float*)take((size_t)BL * D * 4);
  float* kf   = (float*)take((size_t)BL * D * 4);
  float* vf   = (float*)take((size_t)BL * D * 4);
  unsigned short* ctxB = (unsigned short*)take((size_t)BL * D * 2);
  float* attn = (float*)take((size_t)BL * D * 4);
  float* x    = (float*)take((size_t)BL * D * 4);
  unsigned short* xB   = (unsigned short*)take((size_t)BL * D * 2);
  unsigned short* h1B  = (unsigned short*)take((size_t)BL * F * 2);
  float* ff   = (float*)take((size_t)BL * D * 4);

  // src -> bf16 (row-major); weights -> transposed bf16 [N,K]
  {
    int n4 = BL * D / 4;
    cvt_bf16_kernel<<<(n4 + 255) / 256, 256, 0, stream>>>(src, srcB, n4);
  }
  dim3 tblk(256);
  cvt_t_bf16_kernel<<<dim3(D / 32, D / 32), tblk, 0, stream>>>(Wq, WqT, D, D);
  cvt_t_bf16_kernel<<<dim3(D / 32, D / 32), tblk, 0, stream>>>(Wk, WkT, D, D);
  cvt_t_bf16_kernel<<<dim3(D / 32, D / 32), tblk, 0, stream>>>(Wv, WvT, D, D);
  cvt_t_bf16_kernel<<<dim3(D / 32, D / 32), tblk, 0, stream>>>(Wo, WoT, D, D);
  cvt_t_bf16_kernel<<<dim3(F / 32, D / 32), tblk, 0, stream>>>(W1, W1T, D, F);
  cvt_t_bf16_kernel<<<dim3(D / 32, F / 32), tblk, 0, stream>>>(W2, W2T, F, D);

  dim3 gD(D / 64, BL / 128);   // (16,128)
  dim3 gF(F / 64, BL / 128);   // (64,128)

  gemm_bf16_wmma<<<gD, tblk, 0, stream>>>(srcB, WqT, bq, qf, nullptr, BL, D, D, 0);
  gemm_bf16_wmma<<<gD, tblk, 0, stream>>>(srcB, WkT, bk, kf, nullptr, BL, D, D, 0);
  gemm_bf16_wmma<<<gD, tblk, 0, stream>>>(srcB, WvT, bv, vf, nullptr, BL, D, D, 0);

  attn_wmma_kernel<<<256 * 16, tblk, 0, stream>>>(qf, kf, vf, relb, ctxB);

  gemm_bf16_wmma<<<gD, tblk, 0, stream>>>(ctxB, WoT, bo, attn, nullptr, BL, D, D, 0);
  ln_kernel<<<BL, tblk, 0, stream>>>(src, attn, g1, be1, x, xB);
  gemm_bf16_wmma<<<gF, tblk, 0, stream>>>(xB, W1T, b1, nullptr, h1B, BL, F, D, 1);
  gemm_bf16_wmma<<<gD, tblk, 0, stream>>>(h1B, W2T, b2, ff, nullptr, BL, D, F, 0);
  ln_kernel<<<BL, tblk, 0, stream>>>(x, ff, g2, be2, out, nullptr);
}
